// Attention_31112743092529
// MI455X (gfx1250) — compile-verified, matrix operand order checked by reading, not run on hardware
//
#include <hip/hip_runtime.h>

// ---------------------------------------------------------------------------
// MI455X (gfx1250) multi-head attention forward.
// f16 WMMA (16x16x32, f32 accum) for all GEMMs.
// Tensor Data Mover (TDM) double-buffered LDS staging for:
//   - shared weight k-slices in both projection GEMMs (8 waves share B tiles)
//   - K/V key-chunks in the flash-attention stage (4 waves share them)
// ---------------------------------------------------------------------------

typedef __attribute__((ext_vector_type(16))) _Float16     v16h;
typedef __attribute__((ext_vector_type(8)))  float        v8f;
typedef __attribute__((ext_vector_type(4)))  unsigned int u32x4;
typedef __attribute__((ext_vector_type(8)))  unsigned int u32x8;

#define S_LEN      2048
#define DMODEL     1024
#define NHEADS     16
#define HEADDIM    64
#define BATCH      2
#define ROWS_TOTAL (BATCH * S_LEN)   // 4096 flattened (b,s) rows

union V16H {
    v16h     v;
    uint4    u[2];
    _Float16 h[16];
};

// WMMA 16x16x32 f16: D = A*B + C (f32 accum)
__device__ __forceinline__ v8f wmma16(v16h a, v16h b, v8f c) {
    return __builtin_amdgcn_wmma_f32_16x16x32_f16(
        false, a, false, b, (short)0, c, false, false);
}

// Per-lane operand gather for 16-bit A/B operands (ISA 7.12.2 layout):
// lane<16 : K = k0+0..7, k0+16..23 ; lane>=16: K = k0+8..15, k0+24..31
__device__ __forceinline__ v16h load_f16_op(const _Float16* rowbase, int k0, int hi) {
    V16H t;
    t.u[0] = *reinterpret_cast<const uint4*>(rowbase + k0 + hi * 8);
    t.u[1] = *reinterpret_cast<const uint4*>(rowbase + k0 + 16 + hi * 8);
    return t.v;
}

__device__ __forceinline__ v16h load_f32_op(const float* rowbase, int k0, int hi) {
    const float4 a = *reinterpret_cast<const float4*>(rowbase + k0 + hi * 8);
    const float4 b = *reinterpret_cast<const float4*>(rowbase + k0 + hi * 8 + 4);
    const float4 c = *reinterpret_cast<const float4*>(rowbase + k0 + 16 + hi * 8);
    const float4 d = *reinterpret_cast<const float4*>(rowbase + k0 + 16 + hi * 8 + 4);
    V16H t;
    t.h[0]  = (_Float16)a.x; t.h[1]  = (_Float16)a.y; t.h[2]  = (_Float16)a.z; t.h[3]  = (_Float16)a.w;
    t.h[4]  = (_Float16)b.x; t.h[5]  = (_Float16)b.y; t.h[6]  = (_Float16)b.z; t.h[7]  = (_Float16)b.w;
    t.h[8]  = (_Float16)c.x; t.h[9]  = (_Float16)c.y; t.h[10] = (_Float16)c.z; t.h[11] = (_Float16)c.w;
    t.h[12] = (_Float16)d.x; t.h[13] = (_Float16)d.y; t.h[14] = (_Float16)d.z; t.h[15] = (_Float16)d.w;
    return t.v;
}

// LDS byte offset of a generic pointer (low 32 bits of the flat address)
__device__ __forceinline__ unsigned int lds_off(const void* p) {
    return (unsigned int)(unsigned long long)p;
}

// ---------------------------------------------------------------------------
// TDM: async 2D tile load Global -> LDS (CDNA5 ISA).
// 2-operand form (groups 2/3 NULL) = tensors up to 2D. Tracked by TENSORcnt.
// dims/strides in elements (data_size = 2 bytes).
// ---------------------------------------------------------------------------
__device__ __forceinline__ void tdm_load_2d(
    unsigned int lds_addr, const _Float16* gptr,
    unsigned int tensor_d0, unsigned int tensor_d1,
    unsigned long long stride0,
    unsigned int tile_d0, unsigned int tile_d1)
{
    const unsigned long long ga = (unsigned long long)gptr;
    u32x4 g0;
    g0[0] = 1u;                                   // count=1, user descriptor
    g0[1] = lds_addr;                             // LDS byte address
    g0[2] = (unsigned int)ga;                     // global_addr[31:0]
    g0[3] = (unsigned int)(ga >> 32) | (2u << 30);// global_addr[56:32] | type=2

    // group1 bitfields:
    // [17:16]=data_size(1=2B) [79:48]=tensor_dim0 [111:80]=tensor_dim1
    // [127:112]=tile_dim0 [143:128]=tile_dim1 [159:144]=tile_dim2(0)
    // [207:160]=tensor_dim0_stride [255:208]=tensor_dim1_stride(0 for 2D)
    const unsigned long long w0 =
        (unsigned long long)(1u << 16) |
        ((unsigned long long)(tensor_d0 & 0xFFFFu) << 48);
    const unsigned long long w1 =
        (unsigned long long)(tensor_d0 >> 16) |
        ((unsigned long long)tensor_d1 << 16) |
        ((unsigned long long)(tile_d0 & 0xFFFFu) << 48);
    const unsigned long long w2 =
        (unsigned long long)(tile_d1 & 0xFFFFu) |
        ((stride0 & 0xFFFFFFFFull) << 32);
    const unsigned long long w3 = (stride0 >> 32) & 0xFFFFull;

    u32x8 g1;
    g1[0] = (unsigned int)w0; g1[1] = (unsigned int)(w0 >> 32);
    g1[2] = (unsigned int)w1; g1[3] = (unsigned int)(w1 >> 32);
    g1[4] = (unsigned int)w2; g1[5] = (unsigned int)(w2 >> 32);
    g1[6] = (unsigned int)w3; g1[7] = (unsigned int)(w3 >> 32);

    asm volatile("tensor_load_to_lds %0, %1" :: "s"(g0), "s"(g1) : "memory");
}

// ---------------------------------------------------------------------------
// Kernel 0: fp32 -> f16 weight conversion
// ---------------------------------------------------------------------------
__global__ void cvt_f32_f16(const float* __restrict__ src,
                            _Float16* __restrict__ dst, int n) {
    int i = (blockIdx.x * blockDim.x + threadIdx.x) * 4;
    if (i + 3 < n) {
        float4 f = *reinterpret_cast<const float4*>(src + i);
        dst[i + 0] = (_Float16)f.x;
        dst[i + 1] = (_Float16)f.y;
        dst[i + 2] = (_Float16)f.z;
        dst[i + 3] = (_Float16)f.w;
    }
}

// ---------------------------------------------------------------------------
// Kernel 1: QKV projection  out = X @ W^T + b  (grid.z = 0/1/2 -> Q/K/V)
// Block = 8 waves: M=128, N=64. The 64 shared W rows are staged per k-slice
// (64x32 f16 = 4KB) into double-buffered LDS via TDM; all 8 waves consume
// B operands from LDS. A operands (per-wave X rows) convert fp32->f16 inline.
// Q,K stored [b,h,s,dh]; V stored transposed [b,h,dh,s].
// ---------------------------------------------------------------------------
__global__ __launch_bounds__(256) void qkv_proj(
    const float* __restrict__ q_in, const float* __restrict__ k_in,
    const float* __restrict__ v_in,
    const _Float16* __restrict__ Wq16, const _Float16* __restrict__ Wk16,
    const _Float16* __restrict__ Wv16,
    const float* __restrict__ bq, const float* __restrict__ bk,
    const float* __restrict__ bv,
    _Float16* __restrict__ Qh, _Float16* __restrict__ Kh,
    _Float16* __restrict__ Vt)
{
    __shared__ _Float16 wbuf[2][64][32];   // [buf][w-row][k] 8KB

    const int lane = threadIdx.x & 31;
    const int wave = threadIdx.x >> 5;
    const int hi   = lane >> 4;
    const int ln   = lane & 15;
    const int m0   = blockIdx.x * 128 + wave * 16;
    const int n0   = blockIdx.y * 64;
    const int which = blockIdx.z;

    const float*    X;  const _Float16* W;  const float* bias;
    if (which == 0)      { X = q_in; W = Wq16; bias = bq; }
    else if (which == 1) { X = k_in; W = Wk16; bias = bk; }
    else                 { X = v_in; W = Wv16; bias = bv; }

    const float* xrow = X + (size_t)(m0 + ln) * DMODEL;
    const _Float16* wtile = W + (size_t)n0 * DMODEL;   // 64 shared rows

    v8f acc[4];
    #pragma unroll
    for (int t = 0; t < 4; ++t) acc[t] = (v8f){};

    const int NK = DMODEL / 32;   // 32 k-slices

    if (wave == 0) {
        tdm_load_2d(lds_off(&wbuf[0][0][0]), wtile,
                    DMODEL, DMODEL, DMODEL, 32, 64);
    }

    for (int ki = 0; ki < NK; ++ki) {
        const int buf = ki & 1;
        if (wave == 0) {
            if (ki + 1 < NK) {
                tdm_load_2d(lds_off(&wbuf[buf ^ 1][0][0]),
                            wtile + (ki + 1) * 32,
                            DMODEL, DMODEL, DMODEL, 32, 64);
                __builtin_amdgcn_s_wait_tensorcnt(1);  // slice ki landed
            } else {
                __builtin_amdgcn_s_wait_tensorcnt(0);
            }
        }
        __syncthreads();

        v16h a = load_f32_op(xrow, ki * 32, hi);
        #pragma unroll
        for (int t = 0; t < 4; ++t) {
            v16h b = load_f16_op(&wbuf[buf][t * 16 + ln][0], 0, hi);
            acc[t] = wmma16(a, b, acc[t]);
        }

        __syncthreads();   // done with buf before TDM overwrites it
    }

    #pragma unroll
    for (int t = 0; t < 4; ++t) {
        const int   n  = n0 + t * 16 + ln;
        const float bb = bias[n];
        const int   h  = n >> 6;
        const int   dh = n & 63;
        #pragma unroll
        for (int r = 0; r < 8; ++r) {
            const int M     = m0 + r + hi * 8;
            const int b_idx = M >> 11;
            const int s_idx = M & 2047;
            const float val = acc[t][r] + bb;
            if (which == 2) {
                Vt[((size_t)(b_idx * NHEADS + h) * HEADDIM + dh) * S_LEN + s_idx] =
                    (_Float16)val;
            } else {
                _Float16* O = (which == 0) ? Qh : Kh;
                O[((size_t)(b_idx * NHEADS + h) * S_LEN + s_idx) * HEADDIM + dh] =
                    (_Float16)val;
            }
        }
    }
}

// ---------------------------------------------------------------------------
// Kernel 2: flash attention. Block = 4 waves sharing one (b,h); each wave
// owns 16 queries. K/V streamed 32 keys at a time via TDM into
// double-buffered LDS; online softmax; 8 WMMAs per chunk per wave.
// ---------------------------------------------------------------------------
__global__ __launch_bounds__(128) void attention_fa(
    const _Float16* __restrict__ Qh, const _Float16* __restrict__ Kh,
    const _Float16* __restrict__ Vt, _Float16* __restrict__ Ctx)
{
    __shared__ _Float16 kbuf[2][32][HEADDIM]; // [buf][key][dh]   8KB
    __shared__ _Float16 vbuf[2][HEADDIM][32]; // [buf][dh][key]   8KB
    __shared__ _Float16 pbuf[4][16][32];      // per-wave P tiles 4KB

    const int lane = threadIdx.x & 31;
    const int wave = threadIdx.x >> 5;
    const int hi   = lane >> 4;
    const int ln   = lane & 15;

    const int qblk = blockIdx.x & 31;          // 32 q-blocks of 64 queries
    const int hh   = (blockIdx.x >> 5) & 15;
    const int bb   = blockIdx.x >> 9;
    const int q0   = qblk * 64 + wave * 16;

    const _Float16* Qbase = Qh + (size_t)(bb * NHEADS + hh) * S_LEN * HEADDIM;
    const _Float16* Kbase = Kh + (size_t)(bb * NHEADS + hh) * S_LEN * HEADDIM;
    const _Float16* Vbase = Vt + (size_t)(bb * NHEADS + hh) * HEADDIM * S_LEN;

    // Q tile resident as two A operands (Dh = 64 = 2 x K32)
    const _Float16* qrow = Qbase + (size_t)(q0 + ln) * HEADDIM;
    const v16h aq0 = load_f16_op(qrow, 0, hi);
    const v16h aq1 = load_f16_op(qrow, 32, hi);

    v8f o[4];
    #pragma unroll
    for (int t = 0; t < 4; ++t) o[t] = (v8f){};
    float mrow[8], lrow[8];
    #pragma unroll
    for (int r = 0; r < 8; ++r) { mrow[r] = -1.0e30f; lrow[r] = 0.0f; }

    const float scale = 0.125f;   // 1/sqrt(64)
    const int NCHUNK = S_LEN / 32;

    if (wave == 0) {
        tdm_load_2d(lds_off(&kbuf[0][0][0]), Kbase,
                    HEADDIM, S_LEN, HEADDIM, HEADDIM, 32);
        tdm_load_2d(lds_off(&vbuf[0][0][0]), Vbase,
                    S_LEN, HEADDIM, S_LEN, 32, HEADDIM);
    }

    for (int ci = 0; ci < NCHUNK; ++ci) {
        const int buf = ci & 1;
        if (wave == 0) {
            if (ci + 1 < NCHUNK) {
                const int kk = (ci + 1) * 32;
                tdm_load_2d(lds_off(&kbuf[buf ^ 1][0][0]),
                            Kbase + (size_t)kk * HEADDIM,
                            HEADDIM, S_LEN, HEADDIM, HEADDIM, 32);
                tdm_load_2d(lds_off(&vbuf[buf ^ 1][0][0]),
                            Vbase + kk,
                            S_LEN, HEADDIM, S_LEN, 32, HEADDIM);
                __builtin_amdgcn_s_wait_tensorcnt(2);  // chunk ci landed
            } else {
                __builtin_amdgcn_s_wait_tensorcnt(0);
            }
        }
        __syncthreads();   // chunk ci visible to all waves

        const _Float16* kb = &kbuf[buf][0][0];
        const _Float16* vb = &vbuf[buf][0][0];

        // ---- scores: Q(16x64) x K^T(64x32) -> two 16x16 f32 tiles
        v8f c0 = (v8f){}, c1 = (v8f){};
        {
            const _Float16* kr0 = kb + (size_t)ln * HEADDIM;
            c0 = wmma16(aq0, load_f16_op(kr0, 0, hi), c0);
            c0 = wmma16(aq1, load_f16_op(kr0, 32, hi), c0);
            const _Float16* kr1 = kb + (size_t)(16 + ln) * HEADDIM;
            c1 = wmma16(aq0, load_f16_op(kr1, 0, hi), c1);
            c1 = wmma16(aq1, load_f16_op(kr1, 32, hi), c1);
        }

        // ---- online softmax over 32 new columns (rows live across 16 lanes)
        #pragma unroll
        for (int r = 0; r < 8; ++r) {
            float s0 = c0[r] * scale;
            float s1 = c1[r] * scale;
            float vmax = fmaxf(s0, s1);
            #pragma unroll
            for (int mk = 1; mk < 16; mk <<= 1)
                vmax = fmaxf(vmax, __shfl_xor(vmax, mk, 32));
            const float mnew  = fmaxf(mrow[r], vmax);
            const float alpha = __expf(mrow[r] - mnew);
            const float p0 = __expf(s0 - mnew);
            const float p1 = __expf(s1 - mnew);
            float psum = p0 + p1;
            #pragma unroll
            for (int mk = 1; mk < 16; mk <<= 1)
                psum += __shfl_xor(psum, mk, 32);
            lrow[r] = lrow[r] * alpha + psum;
            mrow[r] = mnew;
            o[0][r] *= alpha; o[1][r] *= alpha;
            o[2][r] *= alpha; o[3][r] *= alpha;
            pbuf[wave][r + hi * 8][ln]      = (_Float16)p0;
            pbuf[wave][r + hi * 8][16 + ln] = (_Float16)p1;
        }
        asm volatile("s_wait_dscnt 0" ::: "memory");  // wave-local LDS RAW fence

        // ---- P(16x32) x V(32x64): A from pbuf, B rows from vbuf
        const v16h ap = load_f16_op(&pbuf[wave][ln][0], 0, hi);
        #pragma unroll
        for (int t = 0; t < 4; ++t) {
            v16h bv = load_f16_op(vb + (size_t)(t * 16 + ln) * 32, 0, hi);
            o[t] = wmma16(ap, bv, o[t]);
        }

        __syncthreads();   // all waves done before TDM overwrites buf next iter
    }

    // ---- normalize, emit context [b,s,d] f16
    #pragma unroll
    for (int t = 0; t < 4; ++t) {
        #pragma unroll
        for (int r = 0; r < 8; ++r) {
            const int s_idx = q0 + r + hi * 8;
            const int d     = hh * 64 + t * 16 + ln;
            const float val = o[t][r] / lrow[r];
            Ctx[((size_t)bb * S_LEN + s_idx) * DMODEL + d] = (_Float16)val;
        }
    }
}

// ---------------------------------------------------------------------------
// Kernel 3: output projection  out = Ctx @ Wo^T + bo  (fp32 out)
// Same TDM-staged weight pipeline as qkv_proj; A operands are f16 Ctx rows.
// ---------------------------------------------------------------------------
__global__ __launch_bounds__(256) void out_proj(
    const _Float16* __restrict__ Ctx, const _Float16* __restrict__ Wo16,
    const float* __restrict__ bo, float* __restrict__ out)
{
    __shared__ _Float16 wbuf[2][64][32];   // [buf][w-row][k] 8KB

    const int lane = threadIdx.x & 31;
    const int wave = threadIdx.x >> 5;
    const int hi   = lane >> 4;
    const int ln   = lane & 15;
    const int m0   = blockIdx.x * 128 + wave * 16;
    const int n0   = blockIdx.y * 64;

    const _Float16* arow  = Ctx + (size_t)(m0 + ln) * DMODEL;
    const _Float16* wtile = Wo16 + (size_t)n0 * DMODEL;

    v8f acc[4];
    #pragma unroll
    for (int t = 0; t < 4; ++t) acc[t] = (v8f){};

    const int NK = DMODEL / 32;

    if (wave == 0) {
        tdm_load_2d(lds_off(&wbuf[0][0][0]), wtile,
                    DMODEL, DMODEL, DMODEL, 32, 64);
    }

    for (int ki = 0; ki < NK; ++ki) {
        const int buf = ki & 1;
        if (wave == 0) {
            if (ki + 1 < NK) {
                tdm_load_2d(lds_off(&wbuf[buf ^ 1][0][0]),
                            wtile + (ki + 1) * 32,
                            DMODEL, DMODEL, DMODEL, 32, 64);
                __builtin_amdgcn_s_wait_tensorcnt(1);
            } else {
                __builtin_amdgcn_s_wait_tensorcnt(0);
            }
        }
        __syncthreads();

        v16h a = load_f16_op(arow, ki * 32, hi);
        #pragma unroll
        for (int t = 0; t < 4; ++t) {
            v16h b = load_f16_op(&wbuf[buf][t * 16 + ln][0], 0, hi);
            acc[t] = wmma16(a, b, acc[t]);
        }

        __syncthreads();
    }

    #pragma unroll
    for (int t = 0; t < 4; ++t) {
        const int   n  = n0 + t * 16 + ln;
        const float bb = bo[n];
        #pragma unroll
        for (int r = 0; r < 8; ++r) {
            const int M = m0 + r + hi * 8;
            out[(size_t)M * DMODEL + n] = acc[t][r] + bb;
        }
    }
}

// ---------------------------------------------------------------------------
extern "C" void kernel_launch(void* const* d_in, const int* in_sizes, int n_in,
                              void* d_out, int out_size, void* d_ws, size_t ws_size,
                              hipStream_t stream)
{
    (void)in_sizes; (void)n_in; (void)out_size; (void)ws_size;

    const float* q_in = (const float*)d_in[0];
    const float* k_in = (const float*)d_in[1];
    const float* v_in = (const float*)d_in[2];
    const float* Wq   = (const float*)d_in[3];
    const float* bq   = (const float*)d_in[4];
    const float* Wk   = (const float*)d_in[5];
    const float* bk   = (const float*)d_in[6];
    const float* Wv   = (const float*)d_in[7];
    const float* bv   = (const float*)d_in[8];
    const float* Wo   = (const float*)d_in[9];
    const float* bo   = (const float*)d_in[10];

    _Float16* ws = (_Float16*)d_ws;
    const size_t WE = (size_t)DMODEL * DMODEL;
    const size_t QE = (size_t)ROWS_TOTAL * DMODEL;
    _Float16* Wq16 = ws;
    _Float16* Wk16 = Wq16 + WE;
    _Float16* Wv16 = Wk16 + WE;
    _Float16* Wo16 = Wv16 + WE;
    _Float16* Qh   = Wo16 + WE;
    _Float16* Kh   = Qh + QE;
    _Float16* Vt   = Kh + QE;
    _Float16* Ctx  = Vt + QE;

    {
        const int n = (int)WE;
        const int thr = 256;
        const int blk = (n / 4 + thr - 1) / thr;
        cvt_f32_f16<<<blk, thr, 0, stream>>>(Wq, Wq16, n);
        cvt_f32_f16<<<blk, thr, 0, stream>>>(Wk, Wk16, n);
        cvt_f32_f16<<<blk, thr, 0, stream>>>(Wv, Wv16, n);
        cvt_f32_f16<<<blk, thr, 0, stream>>>(Wo, Wo16, n);
    }

    {
        dim3 grid(ROWS_TOTAL / 128, DMODEL / 64, 3);
        qkv_proj<<<grid, 256, 0, stream>>>(q_in, k_in, v_in,
                                           Wq16, Wk16, Wv16,
                                           bq, bk, bv, Qh, Kh, Vt);
    }

    {
        // 1024 blocks: [b(2)] x [h(16)] x [32 q-blocks of 64 queries]
        attention_fa<<<BATCH * NHEADS * 32, 128, 0, stream>>>(Qh, Kh, Vt, Ctx);
    }

    {
        dim3 grid(ROWS_TOTAL / 128, DMODEL / 64);
        out_proj<<<grid, 256, 0, stream>>>(Ctx, Wo16, bo, (float*)d_out);
    }
}
